// NNBase_12438225289865
// MI455X (gfx1250) — matrix operand order repaired and checked by reading, not compiled, not run on hardware
//
#include <hip/hip_runtime.h>
#include <hip/hip_bf16.h>

// Masked-GRU scan: T=64, N=512, H=2048, A=64.
// MI455X reasoning:
//  - Dominant cost: 64 sequential 512x6144x2048 GEMMs (~825 GFLOP fp32-equiv).
//  - bf16x3 split (Ahi*Bhi + Alo*Bhi + Ahi*Blo) on v_wmma_f32_16x16x32_bf16:
//    ~fp32 accuracy at bf16 matrix-core rate.
//  - w_hh hi/lo planes (~50MB) are L2-resident (192MB). To avoid 32x re-read of
//    weights per step (~1.6GB/step of L2 traffic), each block stages the shared
//    64-col x 32-k weight slab in LDS (double-buffered via
//    GLOBAL_LOAD_ASYNC_TO_LDS_B128 + s_wait_asynccnt) and 8 waves reuse it.
//  - Streaming in/out (~1.5GB @ 23.3TB/s ~ 65us) is negligible.

#define T_DIM 64
#define N_DIM 512
#define H_DIM 2048
#define A_DIM 64
#define G3    (3 * H_DIM)   // 6144

#define BLK_ROWS 128        // 8 waves x 16 rows
#define BLK_COLS 64         // 4 col-tiles of 16
#define COL_STRIDE 80       // LDS bytes/col: 64B k-data + 16B pad (bank permute)
#define PLANE (BLK_COLS * COL_STRIDE)   // 5120 B per hi/lo plane
#define BUFSZ (2 * PLANE)               // hi+lo planes per k-stage

typedef __attribute__((ext_vector_type(16))) __bf16 bf16x16;
typedef __attribute__((ext_vector_type(8)))  __bf16 bf16x8;
typedef __attribute__((ext_vector_type(8)))  float  f32x8;
typedef __attribute__((ext_vector_type(4)))  int    v4i;

// Builtin signature (from hipcc diagnostic): async-to-LDS takes int4 pointers
// in addrspace(1) (global src) and addrspace(3) (LDS dst).
typedef __attribute__((address_space(1))) v4i g_v4i;
typedef __attribute__((address_space(3))) v4i l_v4i;

#if defined(__has_builtin)
#if __has_builtin(__builtin_amdgcn_global_load_async_to_lds_b128)
#define HAVE_ASYNC_LDS 1
#endif
#endif

__device__ __forceinline__ void wait_async_le(void) {
#if defined(HAVE_ASYNC_LDS)
#if __has_builtin(__builtin_amdgcn_s_wait_asynccnt)
  __builtin_amdgcn_s_wait_asynccnt(2);
#else
  asm volatile("s_wait_asynccnt 0x2" ::: "memory");
#endif
#endif
}
__device__ __forceinline__ void wait_async_zero(void) {
#if defined(HAVE_ASYNC_LDS)
#if __has_builtin(__builtin_amdgcn_s_wait_asynccnt)
  __builtin_amdgcn_s_wait_asynccnt(0);
#else
  asm volatile("s_wait_asynccnt 0x0" ::: "memory");
#endif
#endif
}

// ---------------------------------------------------------------------------
// Fragment loaders per CDNA5 WMMA 16-bit VGPR layouts (CDNA5 ISA).
// A (16x32 MxK): lane L<16 holds row M=L, K=+0..7 (v0-3) and +16..23 (v4-7);
// lanes>=16 hold K=+8..15 / +24..31.
__device__ __forceinline__ bf16x16 load_frag_a(const __bf16* p) {
  bf16x8 lo = *reinterpret_cast<const bf16x8*>(p);
  bf16x8 hi = *reinterpret_cast<const bf16x8*>(p + 16);
  return __builtin_shufflevector(lo, hi, 0, 1, 2, 3, 4, 5, 6, 7,
                                         8, 9, 10, 11, 12, 13, 14, 15);
}
// B (32x16 KxN) from LDS stage: lane n<16 = column n holds K=+0..7 (byte 0)
// and +16..23 (byte 32); lanes 16-31 hold K=+8..15 / +24..31 (p at byte 16).
__device__ __forceinline__ bf16x16 load_frag_lds(const char* p) {
  bf16x8 lo = *reinterpret_cast<const bf16x8*>(p);
  bf16x8 hi = *reinterpret_cast<const bf16x8*>(p + 32);
  return __builtin_shufflevector(lo, hi, 0, 1, 2, 3, 4, 5, 6, 7,
                                         8, 9, 10, 11, 12, 13, 14, 15);
}

__device__ __forceinline__ void split_store(float v, __bf16* hi, __bf16* lo, size_t i) {
  __bf16 h = (__bf16)v;                 // RNE: top ~8 mantissa bits
  hi[i] = h;
  lo[i] = (__bf16)(v - (float)h);       // residual: next ~8 bits
}

// ---------------------------------------------------------------------------
__global__ __launch_bounds__(256) void gru_split_weights(
    const float* __restrict__ whh, const float* __restrict__ wih,
    __bf16* __restrict__ whhHi, __bf16* __restrict__ whhLo,
    __bf16* __restrict__ wihHi, __bf16* __restrict__ wihLo) {
  const size_t nHH = (size_t)G3 * H_DIM;
  const size_t nIH = (size_t)G3 * A_DIM;
  size_t i = (size_t)blockIdx.x * blockDim.x + threadIdx.x;
  if (i < nHH) {
    split_store(whh[i], whhHi, whhLo, i);
  } else if (i < nHH + nIH) {
    size_t j = i - nHH;
    split_store(wih[j], wihHi, wihLo, j);
  }
}

// ---------------------------------------------------------------------------
__global__ __launch_bounds__(256) void gru_prep_step(
    const float* __restrict__ h_prev, const float* __restrict__ x_t,
    const float* __restrict__ m_t, const float* __restrict__ pat_t,
    float* __restrict__ hsF32,
    __bf16* __restrict__ hsHi, __bf16* __restrict__ hsLo,
    __bf16* __restrict__ inHi, __bf16* __restrict__ inLo) {
  const size_t nH = (size_t)N_DIM * H_DIM;
  size_t i = (size_t)blockIdx.x * blockDim.x + threadIdx.x;
  if (i < nH) {
    size_t n = i / H_DIM;
    float m = m_t[n];
    float hs = h_prev[i] * m + x_t[i] * (1.0f - m);
    hsF32[i] = hs;
    split_store(hs, hsHi, hsLo, i);
  } else if (i < nH + (size_t)N_DIM * A_DIM) {
    size_t j = i - nH;
    size_t n = j / A_DIM;
    float v = pat_t[j] * m_t[n];
    split_store(v, inHi, inLo, j);
  }
}

// ---------------------------------------------------------------------------
// NT GEMM: C[N_DIM x G3] = A[N_DIM x K] * W[G3 x K]^T, bf16x3, fp32 acc.
// Block: 8 waves = 128 rows x 64 cols. Weight slab (64 cols x 32 k, hi+lo)
// staged in LDS (async double-buffered) and shared by all 8 waves.
__global__ __launch_bounds__(256) void gru_gemm_bf16x3(
    const __bf16* __restrict__ Ahi, const __bf16* __restrict__ Alo,
    const __bf16* __restrict__ Whi, const __bf16* __restrict__ Wlo,
    float* __restrict__ C, int K) {
  __shared__ char smem[2 * BUFSZ];                 // 20 KB

  const int tid  = threadIdx.x;
  const int lane = tid & 31;
  const int wave = tid >> 5;

  const int colGroups = G3 / BLK_COLS;             // 96
  const int rg = blockIdx.x / colGroups;
  const int cg = blockIdx.x - rg * colGroups;
  const int row_base = rg * BLK_ROWS + wave * 16;
  const int col_base = cg * BLK_COLS;

  // Stage-copy mapping: tid -> (column, 16B quarter of its 64B k-slab).
  const int scol = tid >> 2;                       // 0..63
  const int sq   = tid & 3;                        // 0..3
  const __bf16* gHi = Whi + (size_t)(col_base + scol) * K + sq * 8;
  const __bf16* gLo = Wlo + (size_t)(col_base + scol) * K + sq * 8;
  char* sDst = smem + scol * COL_STRIDE + sq * 16;

  // A fragment pointers (per-wave rows, global; L2/WGP$ resident).
  const int lane15 = lane & 15;
  const int half   = lane >> 4;
  const size_t aoff = (size_t)(row_base + lane15) * (size_t)K + (half << 3);
  const __bf16* aHi = Ahi + aoff;
  const __bf16* aLo = Alo + aoff;

  // B fragment LDS byte offset within a buffer (per tile t add t*16 cols).
  const int bOff = lane15 * COL_STRIDE + half * 16;

  f32x8 acc[4] = {f32x8{}, f32x8{}, f32x8{}, f32x8{}};

  auto stage = [&](int kb, int buf) {
    char* d = sDst + buf * BUFSZ;
#if defined(HAVE_ASYNC_LDS)
    __builtin_amdgcn_global_load_async_to_lds_b128(
        (g_v4i*)(gHi + kb), (l_v4i*)d, 0, 0);
    __builtin_amdgcn_global_load_async_to_lds_b128(
        (g_v4i*)(gLo + kb), (l_v4i*)(d + PLANE), 0, 0);
#else
    *reinterpret_cast<bf16x8*>(d)         = *reinterpret_cast<const bf16x8*>(gHi + kb);
    *reinterpret_cast<bf16x8*>(d + PLANE) = *reinterpret_cast<const bf16x8*>(gLo + kb);
#endif
  };

  stage(0, 0);
  const int nk = K >> 5;
  for (int i = 0; i < nk; ++i) {
    const int kb = i << 5;
    const bool more = (i + 1 < nk);
    if (more) {
      stage(kb + 32, (i + 1) & 1);
      wait_async_le();                 // retire stage i (async ops in-order)
      __builtin_prefetch(aHi + kb + 32, 0, 3);
    } else {
      wait_async_zero();
    }
    __syncthreads();                   // stage i visible to all waves

    const char* sbuf = smem + (i & 1) * BUFSZ;
    bf16x16 ah = load_frag_a(aHi + kb);
    bf16x16 al = load_frag_a(aLo + kb);
#pragma unroll
    for (int t = 0; t < 4; ++t) {
      const char* bp = sbuf + t * 16 * COL_STRIDE + bOff;
      bf16x16 bh = load_frag_lds(bp);
      bf16x16 bl = load_frag_lds(bp + PLANE);
      acc[t] = __builtin_amdgcn_wmma_f32_16x16x32_bf16(
          false, ah, false, bh, (short)0, acc[t], false, false);
      acc[t] = __builtin_amdgcn_wmma_f32_16x16x32_bf16(
          false, al, false, bh, (short)0, acc[t], false, false);
      acc[t] = __builtin_amdgcn_wmma_f32_16x16x32_bf16(
          false, ah, false, bl, (short)0, acc[t], false, false);
    }
    __syncthreads();                   // protect buffer before next overwrite
  }

  // C/D layout: VGPR i -> row M = half*8 + i, col N = lane&15.
  const int crow = row_base + (half << 3);
#pragma unroll
  for (int t = 0; t < 4; ++t) {
    float* cp = C + (size_t)crow * G3 + (col_base + t * 16 + lane15);
#pragma unroll
    for (int i = 0; i < 8; ++i) cp[(size_t)i * G3] = acc[t][i];
  }
}

// ---------------------------------------------------------------------------
__global__ __launch_bounds__(256) void gru_gates_step(
    const float* __restrict__ Ghh, const float* __restrict__ Gih,
    const float* __restrict__ b_ih, const float* __restrict__ b_hh,
    const float* __restrict__ hsF32, const float* __restrict__ x_t,
    const float* __restrict__ att_t,
    float* __restrict__ pmu_out, float* __restrict__ attx_out) {
  size_t i = (size_t)blockIdx.x * blockDim.x + threadIdx.x;
  if (i >= (size_t)N_DIM * H_DIM) return;
  size_t n  = i / H_DIM;
  size_t hc = i - n * H_DIM;
  size_t g0 = n * G3 + hc;

  float gr  = Ghh[g0]             + Gih[g0]             + b_ih[hc]            + b_hh[hc];
  float gz  = Ghh[g0 + H_DIM]     + Gih[g0 + H_DIM]     + b_ih[hc + H_DIM]    + b_hh[hc + H_DIM];
  float i_n = Gih[g0 + 2 * H_DIM] + b_ih[hc + 2 * H_DIM];
  float h_n = Ghh[g0 + 2 * H_DIM] + b_hh[hc + 2 * H_DIM];

  float r  = 1.0f / (1.0f + __expf(-gr));
  float z  = 1.0f / (1.0f + __expf(-gz));
  float nn = tanhf(i_n + r * h_n);

  float hs  = hsF32[i];
  float pmu = (1.0f - z) * nn + z * hs;
  float a   = att_t[i];
  float ax  = x_t[i] * a + pmu * (1.0f - a);

  pmu_out[i]  = pmu;
  attx_out[i] = ax;
}

// ---------------------------------------------------------------------------
extern "C" void kernel_launch(void* const* d_in, const int* in_sizes, int n_in,
                              void* d_out, int out_size, void* d_ws, size_t ws_size,
                              hipStream_t stream) {
  const float* x   = (const float*)d_in[0];   // [T*N, H]
  const float* hxs = (const float*)d_in[1];   // [N, H]
  const float* att = (const float*)d_in[2];   // [T*N, H]
  const float* msk = (const float*)d_in[3];   // [T*N, 1]
  const float* pat = (const float*)d_in[4];   // [T, N, A]
  const float* wih = (const float*)d_in[5];   // [3H, A]
  const float* whh = (const float*)d_in[6];   // [3H, H]
  const float* bih = (const float*)d_in[7];   // [3H]
  const float* bhh = (const float*)d_in[8];   // [3H]

  const size_t NH  = (size_t)N_DIM * H_DIM;
  const size_t TNH = (size_t)T_DIM * NH;
  float* out_pmu = (float*)d_out;
  float* out_att = out_pmu + TNH;
  float* out_h   = out_att + TNH;

  // Workspace carve-out (~86 MB).
  char* ws = (char*)d_ws;
  size_t off = 0;
  auto take = [&](size_t bytes) -> char* {
    char* p = ws + off;
    off = (off + bytes + 255) & ~(size_t)255;
    return p;
  };
  __bf16* whhHi = (__bf16*)take((size_t)G3 * H_DIM * sizeof(__bf16));
  __bf16* whhLo = (__bf16*)take((size_t)G3 * H_DIM * sizeof(__bf16));
  __bf16* wihHi = (__bf16*)take((size_t)G3 * A_DIM * sizeof(__bf16));
  __bf16* wihLo = (__bf16*)take((size_t)G3 * A_DIM * sizeof(__bf16));
  __bf16* hsHi  = (__bf16*)take(NH * sizeof(__bf16));
  __bf16* hsLo  = (__bf16*)take(NH * sizeof(__bf16));
  float*  hsF32 = (float*) take(NH * sizeof(float));
  __bf16* inHi  = (__bf16*)take((size_t)N_DIM * A_DIM * sizeof(__bf16));
  __bf16* inLo  = (__bf16*)take((size_t)N_DIM * A_DIM * sizeof(__bf16));
  float*  Ghh   = (float*) take((size_t)N_DIM * G3 * sizeof(float));
  float*  Gih   = (float*) take((size_t)N_DIM * G3 * sizeof(float));
  (void)ws_size; (void)in_sizes; (void)n_in; (void)out_size;

  // 1) Weight splits.
  {
    size_t total = (size_t)G3 * H_DIM + (size_t)G3 * A_DIM;
    int blocks = (int)((total + 255) / 256);
    gru_split_weights<<<blocks, 256, 0, stream>>>(whh, wih, whhHi, whhLo, wihHi, wihLo);
  }

  const int prepBlocks = (int)((NH + (size_t)N_DIM * A_DIM + 255) / 256);
  const int gateBlocks = (int)((NH + 255) / 256);
  const int gemmBlocks = (N_DIM / BLK_ROWS) * (G3 / BLK_COLS);   // 4*96 = 384

  // 2) Sequential scan; carry h(t+1) == attended_x(t) in-place in d_out.
  for (int t = 0; t < T_DIM; ++t) {
    const float* h_prev = (t == 0) ? hxs : (out_att + (size_t)(t - 1) * NH);
    const float* x_t    = x   + (size_t)t * NH;
    const float* a_t    = att + (size_t)t * NH;
    const float* m_t    = msk + (size_t)t * N_DIM;
    const float* pat_t  = pat + (size_t)t * N_DIM * A_DIM;

    gru_prep_step<<<prepBlocks, 256, 0, stream>>>(
        h_prev, x_t, m_t, pat_t, hsF32, hsHi, hsLo, inHi, inLo);

    gru_gemm_bf16x3<<<gemmBlocks, 256, 0, stream>>>(
        hsHi, hsLo, whhHi, whhLo, Ghh, H_DIM);
    gru_gemm_bf16x3<<<gemmBlocks, 256, 0, stream>>>(
        inHi, inLo, wihHi, wihLo, Gih, A_DIM);

    gru_gates_step<<<gateBlocks, 256, 0, stream>>>(
        Ghh, Gih, bih, bhh, hsF32, x_t, a_t,
        out_pmu + (size_t)t * NH, out_att + (size_t)t * NH);
  }

  // 3) h_final == attended_x at t = T-1.
  (void)hipMemcpyAsync(out_h, out_att + (size_t)(T_DIM - 1) * NH,
                       NH * sizeof(float), hipMemcpyDeviceToDevice, stream);
}
